// GNet_33698313404770
// MI455X (gfx1250) — matrix operand order repaired and checked by reading, not compiled, not run on hardware
//
#include <hip/hip_runtime.h>
#include <hip/hip_bf16.h>
#include <cstdint>

// ---------------------------------------------------------------------------
// CDNA5 (gfx1250) implementation of the temporal-graph attention layer.
// Dense math runs on v_wmma_f32_16x16x32_bf16 (fp32 accumulate); the typed-
// linear basis GEMMs are hoisted from edges (E=80k) to nodes (N=20k) using
// linearity of LN/typed_linear, a 4x FLOP cut. GEMM tiles are staged with
// gfx1250 async global->LDS copies (ASYNCcnt) in a double-buffered pipeline.
// Edge stages are wave-per-edge bandwidth kernels using f32 global atomics.
// ---------------------------------------------------------------------------

typedef __attribute__((ext_vector_type(16))) __bf16 v16bf;
typedef __attribute__((ext_vector_type(8)))  float  v8f;

__device__ __forceinline__ unsigned short f2bf(float x) {
    unsigned u = __float_as_uint(x);
    unsigned r = (u + 0x7fffu + ((u >> 16) & 1u)) >> 16;   // round-nearest-even
    return (unsigned short)r;
}
__device__ __forceinline__ float bf2f(unsigned short h) {
    return __uint_as_float(((unsigned)h) << 16);
}

// Async 16B global->LDS copy (gfx1250, tracked by ASYNCcnt). `lds` is the LDS
// byte offset (low 32 bits of a flat shared pointer: the LDS aperture only
// occupies addr[63:32], so the low half is the raw LDS address).
__device__ __forceinline__ void async_copy_b128(unsigned lds, const void* gaddr) {
    asm volatile("global_load_async_to_lds_b128 %0, %1, off"
                 :: "v"(lds), "v"(gaddr) : "memory");
}
__device__ __forceinline__ unsigned lds_off(const void* p) {
    return (unsigned)(unsigned long long)p;
}

// ------------------------------- WMMA GEMM ---------------------------------
// C[M,N] = A[M,K] (bf16 row-major) x B[K,N], with B provided transposed:
// Bt[N,K] bf16 row-major. Block tile 128x128x32, 256 threads = 8 waves,
// wave grid 4(M) x 2(N), each wave owns 32x64 = 2x4 WMMA tiles.
// Tiles are double-buffered in LDS and filled with async copies; each
// iteration issues the next tile (4 async ops/wave) and waits asynccnt<=4 so
// only the current tile's transfers must have landed.
#define BM 128
#define BN 128
#define BK 32
#define LDSW 40   // padded row stride (bf16 elems): 80B -> conflict-free b128 reads

template <bool OUT_BF16>
__global__ __launch_bounds__(256) void gemm_bf16_wmma(
    const unsigned short* __restrict__ A, const unsigned short* __restrict__ Bt,
    float* __restrict__ Cf, unsigned short* __restrict__ Cb, int M, int N, int K)
{
    __shared__ __align__(16) unsigned short lds_a[2][BM * LDSW];
    __shared__ __align__(16) unsigned short lds_b[2][BN * LDSW];

    const int tid  = threadIdx.x;
    const int wave = tid >> 5;
    const int lane = tid & 31;
    const int wm   = wave & 3;      // 4 waves along M
    const int wn   = wave >> 2;     // 2 waves along N
    const int half = lane >> 4;     // lane half selects K sub-block (ISA layout)
    const int lr   = lane & 15;

    const int block_m = blockIdx.y * BM;
    const int block_n = blockIdx.x * BN;

    // Per-thread staging slots: 2 passes x (1 A + 1 B) 16B chunk.
    int s_row[2], s_c8[2], s_ga[2];
    const unsigned la0 = lds_off(&lds_a[0][0]);
    const unsigned lb0 = lds_off(&lds_b[0][0]);
#pragma unroll
    for (int p = 0; p < 2; ++p) {
        int idx = p * 256 + tid;
        s_row[p] = idx >> 2;
        s_c8[p]  = idx & 3;
        int gr = block_m + s_row[p];
        s_ga[p] = gr > M - 1 ? M - 1 : gr;          // row clamp for A
    }

    auto issue_tile = [&](int buf, int k0) {
#pragma unroll
        for (int p = 0; p < 2; ++p) {
            const int row = s_row[p], c8 = s_c8[p];
            const unsigned loff = (unsigned)(buf * BM * LDSW + row * LDSW + c8 * 8) * 2u;
            async_copy_b128(la0 + loff, A  + (size_t)s_ga[p] * K + k0 + c8 * 8);
            async_copy_b128(lb0 + loff, Bt + (size_t)(block_n + row) * K + k0 + c8 * 8);
        }
    };

    v8f acc[2][4];
    for (int i = 0; i < 2; ++i)
        for (int j = 0; j < 4; ++j)
            for (int e = 0; e < 8; ++e) acc[i][j][e] = 0.f;

    const int KT = K / BK;
    issue_tile(0, 0);                       // prologue: tile 0 in flight

    for (int kt = 0; kt < KT; ++kt) {
        const int cur = kt & 1;
        if (kt + 1 < KT) {
            issue_tile(cur ^ 1, (kt + 1) * BK);   // prefetch next tile
            asm volatile("s_wait_asynccnt 0x4" ::: "memory");  // cur tile landed
        } else {
            asm volatile("s_wait_asynccnt 0x0" ::: "memory");
        }
        __syncthreads();                    // all waves' copies visible

        union Frag { uint4 u[2]; v16bf v; };
        Frag afrag[2], bfrag[4];
        // A fragment 16x32 (ISA 7.12.2): lanes 0-15 K=0..7/16..23, 16-31 K=8..15/24..31
#pragma unroll
        for (int am = 0; am < 2; ++am) {
            const unsigned short* p =
                &lds_a[cur][(wm * 32 + am * 16 + lr) * LDSW + half * 8];
            afrag[am].u[0] = *(const uint4*)(p);        // K = half*8 .. +7
            afrag[am].u[1] = *(const uint4*)(p + 16);   // K = 16+half*8 .. +7
        }
        // B fragment 32x16: lanes 0-15 col n, K=0..7/16..23; lanes 16-31 K=8..15/24..31
#pragma unroll
        for (int bn = 0; bn < 4; ++bn) {
            const unsigned short* p =
                &lds_b[cur][(wn * 64 + bn * 16 + lr) * LDSW + half * 8];
            bfrag[bn].u[0] = *(const uint4*)(p);
            bfrag[bn].u[1] = *(const uint4*)(p + 16);
        }
#pragma unroll
        for (int am = 0; am < 2; ++am)
#pragma unroll
            for (int bn = 0; bn < 4; ++bn)
                acc[am][bn] = __builtin_amdgcn_wmma_f32_16x16x32_bf16(
                    false, afrag[am].v, false, bfrag[bn].v,
                    (short)0, acc[am][bn], false, false);
        __syncthreads();                    // done reading cur -> safe to refill
    }

    // C layout: lanes 0-15 M=i, lanes 16-31 M=8+i, col = lane&15.
#pragma unroll
    for (int am = 0; am < 2; ++am)
#pragma unroll
        for (int bn = 0; bn < 4; ++bn) {
            int col = block_n + wn * 64 + bn * 16 + lr;
#pragma unroll
            for (int i = 0; i < 8; ++i) {
                int row = block_m + wm * 32 + am * 16 + half * 8 + i;
                if (row < M) {
                    if (OUT_BF16) Cb[(size_t)row * N + col] = f2bf(acc[am][bn][i]);
                    else          Cf[(size_t)row * N + col] = acc[am][bn][i];
                }
            }
        }
}

// ----------------------------- support kernels -----------------------------
__global__ void f32_to_bf16(const float* __restrict__ x,
                            unsigned short* __restrict__ y, long n) {
    long i = (long)blockIdx.x * 256 + threadIdx.x;
    long stride = (long)gridDim.x * 256;
    for (; i < n; i += stride) y[i] = f2bf(x[i]);
}

__global__ void transpose_bf16(const float* __restrict__ x,
                               unsigned short* __restrict__ y, int D) {
    long i = (long)blockIdx.x * 256 + threadIdx.x;
    if (i < (long)D * D) {
        int n = (int)(i / D), k = (int)(i % D);
        y[(size_t)n * D + k] = f2bf(x[(size_t)k * D + n]);
    }
}

__global__ __launch_bounds__(256) void layernorm_bf16(
    const float* __restrict__ X, const float* __restrict__ g,
    const float* __restrict__ b, unsigned short* __restrict__ Y, int D) {
    __shared__ float s1[256], s2[256];
    int r = blockIdx.x, t = threadIdx.x;
    const float* x = X + (size_t)r * D;
    float a0 = x[t], a1 = x[t + 256], a2 = x[t + 512];
    s1[t] = a0 + a1 + a2;
    s2[t] = a0 * a0 + a1 * a1 + a2 * a2;
    __syncthreads();
    for (int off = 128; off > 0; off >>= 1) {
        if (t < off) { s1[t] += s1[t + off]; s2[t] += s2[t + off]; }
        __syncthreads();
    }
    float mean = s1[0] * (1.f / 768.f);
    float var  = s2[0] * (1.f / 768.f) - mean * mean;
    float rstd = rsqrtf(var + 1e-5f);
    unsigned short* y = Y + (size_t)r * D;
    y[t]       = f2bf((a0 - mean) * rstd * g[t]       + b[t]);
    y[t + 256] = f2bf((a1 - mean) * rstd * g[t + 256] + b[t + 256]);
    y[t + 512] = f2bf((a2 - mean) * rstd * g[t + 512] + b[t + 512]);
}

__global__ void seg_init(float* hout, float* den, int* mmax, long nd, int n) {
    long i = (long)blockIdx.x * 256 + threadIdx.x;
    long stride = (long)gridDim.x * 256;
    for (long j = i; j < nd; j += stride) hout[j] = 0.f;
    for (long j = i; j < n;  j += stride) { den[j] = 0.f; mmax[j] = (int)0x80000000; }
}

// Wave per edge: build message (time-encode + relu(typed combine)), build the
// attention key (raw n_dst gather, or relu'd typed combine for homo branch),
// dot-reduce, and atomic-max the score into the segment maximum.
__global__ __launch_bounds__(256) void edge_score(
    const unsigned short* __restrict__ Y0, const unsigned short* __restrict__ Y1,
    const unsigned short* __restrict__ Y2, const float* __restrict__ coeff_msg,
    const int* __restrict__ et_msg,
    const unsigned short* __restrict__ K0, const unsigned short* __restrict__ K1,
    const unsigned short* __restrict__ K2, const float* __restrict__ coeff_key,
    const int* __restrict__ et_key, const float* __restrict__ ndst,
    const float* __restrict__ tA, const float* __restrict__ tB,
    const float* __restrict__ freq, const float* __restrict__ phase,
    const int* __restrict__ src, const int* __restrict__ dst,
    unsigned short* __restrict__ msg_out, float* __restrict__ esc,
    int* __restrict__ mmax, int E, int D)
{
    int wave = threadIdx.x >> 5, lane = threadIdx.x & 31;
    int e = blockIdx.x * 8 + wave;
    if (e >= E) return;
    int s = src[e], dd = dst[e];
    int em = et_msg[e];
    float c0 = coeff_msg[em * 3 + 0], c1 = coeff_msg[em * 3 + 1],
          c2 = coeff_msg[em * 3 + 2];
    float k0c = 0.f, k1c = 0.f, k2c = 0.f;
    if (K0) {
        int ek = et_key[e];
        k0c = coeff_key[ek * 3 + 0]; k1c = coeff_key[ek * 3 + 1];
        k2c = coeff_key[ek * 3 + 2];
    }
    float t = fmaxf(tA[e], tB[e]);
    size_t sb = (size_t)s * D, db = (size_t)dd * D, eb = (size_t)e * D;
    float dot = 0.f;
    for (int d = lane; d < D; d += 32) {
        float m = c0 * bf2f(Y0[sb + d]) + c1 * bf2f(Y1[sb + d]) +
                  c2 * bf2f(Y2[sb + d]);
        m = fmaxf(m, 0.f);
        m += __cosf(t * freq[d] + phase[d]);
        msg_out[eb + d] = f2bf(m);
        float kv;
        if (K0) kv = fmaxf(k0c * bf2f(K0[db + d]) + k1c * bf2f(K1[db + d]) +
                           k2c * bf2f(K2[db + d]), 0.f);
        else    kv = ndst[db + d];
        dot += m * kv;
    }
    for (int off = 16; off; off >>= 1) dot += __shfl_xor(dot, off, 32);
    if (lane == 0) {
        float sc = dot * 0.03608439182435161f;  // 1/sqrt(768)
        esc[e] = sc;
        int ix = __float_as_int(sc);
        ix = ix >= 0 ? ix : (ix ^ 0x7fffffff);  // monotone float->int
        atomicMax(&mmax[dd], ix);
    }
}

__global__ void edge_expsum(const float* __restrict__ esc,
                            const int* __restrict__ dst,
                            const int* __restrict__ mmax,
                            float* __restrict__ e2, float* __restrict__ den,
                            int E) {
    int e = blockIdx.x * 256 + threadIdx.x;
    if (e >= E) return;
    int d = dst[e];
    int ix = mmax[d];
    float m = __int_as_float(ix >= 0 ? ix : (ix ^ 0x7fffffff));
    float ex = __expf(esc[e] - m);
    e2[e] = ex;
    unsafeAtomicAdd(&den[d], ex);
}

__global__ __launch_bounds__(256) void edge_scatter(
    const unsigned short* __restrict__ msg, const float* __restrict__ e2,
    const float* __restrict__ den, const int* __restrict__ dst,
    float* __restrict__ hout, int E, int D) {
    int wave = threadIdx.x >> 5, lane = threadIdx.x & 31;
    int e = blockIdx.x * 8 + wave;
    if (e >= E) return;
    int dd = dst[e];
    float dn = den[dd];
    float a = e2[e] / (dn == 0.f ? 1.f : dn);
    size_t eb = (size_t)e * D, db = (size_t)dd * D;
    for (int d = lane; d < D; d += 32)
        unsafeAtomicAdd(&hout[db + d], a * bf2f(msg[eb + d]));
}

__global__ void concat_bf16(const float* __restrict__ a,
                            const float* __restrict__ b,
                            unsigned short* __restrict__ y, int N, int D) {
    long tot = (long)N * D;
    long i = (long)blockIdx.x * 256 + threadIdx.x;
    long stride = (long)gridDim.x * 256;
    for (long j = i; j < tot; j += stride) {
        long r = j / D, d = j % D;
        y[r * 2 * D + d]     = f2bf(a[j]);
        y[r * 2 * D + D + d] = f2bf(b[j]);
    }
}

__global__ __launch_bounds__(256) void finalize_norm(
    const float* __restrict__ z, const float* __restrict__ bias,
    float* __restrict__ out, int D) {
    __shared__ float s[256];
    int r = blockIdx.x, t = threadIdx.x;
    const float* x = z + (size_t)r * D;
    float v0 = fmaxf(x[t] + bias[t], 0.f);
    float v1 = fmaxf(x[t + 256] + bias[t + 256], 0.f);
    float v2 = fmaxf(x[t + 512] + bias[t + 512], 0.f);
    s[t] = v0 * v0 + v1 * v1 + v2 * v2;
    __syncthreads();
    for (int off = 128; off > 0; off >>= 1) {
        if (t < off) s[t] += s[t + off];
        __syncthreads();
    }
    float nrm = sqrtf(s[0]);
    float inv = (nrm == 0.f) ? 1.f : 1.f / nrm;
    float* o = out + (size_t)r * D;
    o[t] = v0 * inv; o[t + 256] = v1 * inv; o[t + 512] = v2 * inv;
}

// --------------------------------- driver ----------------------------------
extern "C" void kernel_launch(void* const* d_in, const int* in_sizes, int n_in,
                              void* d_out, int out_size, void* d_ws,
                              size_t ws_size, hipStream_t stream) {
    (void)n_in; (void)out_size; (void)ws_size;
    const int D = 768;
    const float* h          = (const float*)d_in[0];
    const float* t_hete     = (const float*)d_in[1];
    const float* t1_homo    = (const float*)d_in[2];
    const float* t2_homo    = (const float*)d_in[3];
    const float* w_src      = (const float*)d_in[4];
    const float* w_dst      = (const float*)d_in[5];
    const float* ln_g       = (const float*)d_in[6];
    const float* ln_b       = (const float*)d_in[7];
    const float* src_bases  = (const float*)d_in[8];
    const float* src_coeff  = (const float*)d_in[9];
    const float* dst_bases  = (const float*)d_in[10];
    const float* dst_coeff  = (const float*)d_in[11];
    const float* user_bases = (const float*)d_in[12];
    const float* user_coeff = (const float*)d_in[13];
    const float* lin_w      = (const float*)d_in[14];
    const float* lin_b      = (const float*)d_in[15];
    const float* time_freq  = (const float*)d_in[16];
    const float* time_phase = (const float*)d_in[17];
    const int* src_hete     = (const int*)d_in[18];
    const int* dst_hete     = (const int*)d_in[19];
    const int* et_hete      = (const int*)d_in[20];
    const int* src_homo     = (const int*)d_in[21];
    const int* dst_homo     = (const int*)d_in[22];
    const int* et1_homo     = (const int*)d_in[23];
    const int* et2_homo     = (const int*)d_in[24];

    const int N = in_sizes[0] / D;
    const int E = in_sizes[1];

    char* ws = (char*)d_ws;
    size_t off = 0;
    auto carve = [&](size_t bytes) -> char* {
        char* p = ws + off;
        off = (off + bytes + 255) & ~(size_t)255;
        return p;
    };
    const size_t ND = (size_t)N * D, DD = (size_t)D * D, ED = (size_t)E * D;
    unsigned short* hb    = (unsigned short*)carve(ND * 2);
    unsigned short* wsb   = (unsigned short*)carve(DD * 2);     // w_src rows == Bt
    unsigned short* wdb   = (unsigned short*)carve(DD * 2);
    unsigned short* lwb   = (unsigned short*)carve((size_t)D * 2 * D * 2);
    unsigned short* basesb = (unsigned short*)carve(9 * DD * 2); // src0-2,dst0-2,user0-2 (transposed)
    float*          nsrc  = (float*)carve(ND * 4);
    float*          ndst  = (float*)carve(ND * 4);
    unsigned short* lnsrc = (unsigned short*)carve(ND * 2);
    unsigned short* lndst = (unsigned short*)carve(ND * 2);
    unsigned short* Yb[3], *Zb[3], *Ub[3];
    for (int i = 0; i < 3; ++i) Yb[i] = (unsigned short*)carve(ND * 2);
    for (int i = 0; i < 3; ++i) Zb[i] = (unsigned short*)carve(ND * 2);
    for (int i = 0; i < 3; ++i) Ub[i] = (unsigned short*)carve(ND * 2);
    unsigned short* msg  = (unsigned short*)carve(ED * 2);
    float* esc  = (float*)carve((size_t)E * 4);
    float* e2   = (float*)carve((size_t)E * 4);
    float* den  = (float*)carve((size_t)N * 4);
    int*   mmax = (int*)carve((size_t)N * 4);
    float* hhet = (float*)carve(ND * 4);
    float* hhom = (float*)carve(ND * 4);
    unsigned short* ncat = (unsigned short*)carve(ND * 2 * 2);
    float* zbuf = (float*)carve(ND * 4);

    // ---- prep: bf16 conversion / weight transposition ----
    f32_to_bf16<<<2048, 256, 0, stream>>>(h, hb, (long)ND);
    f32_to_bf16<<<512, 256, 0, stream>>>(w_src, wsb, (long)DD);   // h@W.T: Bt == W
    f32_to_bf16<<<512, 256, 0, stream>>>(w_dst, wdb, (long)DD);
    f32_to_bf16<<<512, 256, 0, stream>>>(lin_w, lwb, (long)D * 2 * D);
    int tgrid = (int)((DD + 255) / 256);
    for (int i = 0; i < 3; ++i) {
        transpose_bf16<<<tgrid, 256, 0, stream>>>(src_bases + i * DD,  basesb + (0 + i) * DD, D);
        transpose_bf16<<<tgrid, 256, 0, stream>>>(dst_bases + i * DD,  basesb + (3 + i) * DD, D);
        transpose_bf16<<<tgrid, 256, 0, stream>>>(user_bases + i * DD, basesb + (6 + i) * DD, D);
    }

    dim3 gN(D / BN, (N + BM - 1) / BM);
    // ---- node projections ----
    gemm_bf16_wmma<false><<<gN, 256, 0, stream>>>(hb, wsb, nsrc, nullptr, N, D, D);
    gemm_bf16_wmma<false><<<gN, 256, 0, stream>>>(hb, wdb, ndst, nullptr, N, D, D);
    layernorm_bf16<<<N, 256, 0, stream>>>(nsrc, ln_g, ln_b, lnsrc, D);
    layernorm_bf16<<<N, 256, 0, stream>>>(ndst, ln_g, ln_b, lndst, D);
    // ---- hoisted typed-linear basis GEMMs (per node, not per edge) ----
    for (int i = 0; i < 3; ++i) {
        gemm_bf16_wmma<true><<<gN, 256, 0, stream>>>(lnsrc, basesb + (0 + i) * DD, nullptr, Yb[i], N, D, D);
        gemm_bf16_wmma<true><<<gN, 256, 0, stream>>>(lndst, basesb + (3 + i) * DD, nullptr, Zb[i], N, D, D);
        gemm_bf16_wmma<true><<<gN, 256, 0, stream>>>(lnsrc, basesb + (6 + i) * DD, nullptr, Ub[i], N, D, D);
    }

    int egrid  = (E + 7) / 8;
    int egrid1 = (E + 255) / 256;
    // ---- heterogeneous branch ----
    seg_init<<<2048, 256, 0, stream>>>(hhet, den, mmax, (long)ND, N);
    edge_score<<<egrid, 256, 0, stream>>>(
        Yb[0], Yb[1], Yb[2], src_coeff, et_hete,
        nullptr, nullptr, nullptr, nullptr, nullptr, ndst,
        t_hete, t_hete, time_freq, time_phase, src_hete, dst_hete,
        msg, esc, mmax, E, D);
    edge_expsum<<<egrid1, 256, 0, stream>>>(esc, dst_hete, mmax, e2, den, E);
    edge_scatter<<<egrid, 256, 0, stream>>>(msg, e2, den, dst_hete, hhet, E, D);
    // ---- homogeneous branch ----
    seg_init<<<2048, 256, 0, stream>>>(hhom, den, mmax, (long)ND, N);
    edge_score<<<egrid, 256, 0, stream>>>(
        Ub[0], Ub[1], Ub[2], user_coeff, et2_homo,
        Zb[0], Zb[1], Zb[2], dst_coeff, et1_homo, ndst,
        t1_homo, t2_homo, time_freq, time_phase, src_homo, dst_homo,
        msg, esc, mmax, E, D);
    edge_expsum<<<egrid1, 256, 0, stream>>>(esc, dst_homo, mmax, e2, den, E);
    edge_scatter<<<egrid, 256, 0, stream>>>(msg, e2, den, dst_homo, hhom, E, D);

    // ---- output head: concat -> linear (K=1536) -> relu -> L2 normalize ----
    concat_bf16<<<2048, 256, 0, stream>>>(hhet, hhom, ncat, N, D);
    gemm_bf16_wmma<false><<<gN, 256, 0, stream>>>(ncat, lwb, zbuf, nullptr, N, D, 2 * D);
    finalize_norm<<<N, 256, 0, stream>>>(zbuf, lin_b, (float*)d_out, D);
}
